// ESNCell_57664230916639
// MI455X (gfx1250) — matrix operand order repaired and checked by reading, not compiled, not run on hardware
//
#include <hip/hip_runtime.h>

// ---- types ------------------------------------------------------------
typedef __attribute__((ext_vector_type(16))) __bf16 v16bf;
typedef __attribute__((ext_vector_type(8)))  __bf16 v8bf;
typedef __attribute__((ext_vector_type(8)))  float  v8f;
typedef __attribute__((ext_vector_type(4)))  float  v4f;

// ---- problem constants ------------------------------------------------
#define T_STEPS  4096
#define BATCH    64
#define IN_D     128
#define HID      1024
#define NWG      16     // workgroups; each owns HID/NWG = 64 output columns
#define HS       64     // hidden-slice width per WG
#define NTHREADS 512    // 16 wave32s -> 4x4 grid of 16x16 output tiles

// ---- LDS layout (bytes) ----------------------------------------------
#define WHH_OFF   0
#define WIH_OFF   (HS * HID * 2)                 // 131072
#define HL_OFF    (WIH_OFF + HS * IN_D * 2)      // 147456
#define BIAS_OFF  (HL_OFF + BATCH * HID * 2)     // 278528
#define LDS_BYTES (BIAS_OFF + HS * 4)            // 278784  (< 320KB WGP LDS)

// ---- workspace layout (bytes) -----------------------------------------
#define CTR_BYTES (T_STEPS * 4)                  // per-step barrier counters
#define HA_OFF    CTR_BYTES                      // 16384
#define HB_OFF    (HA_OFF + BATCH * HID * 2)     // 147456
// total ws use: 278528 bytes

__global__ void esn_init_kernel(unsigned* __restrict__ p, int n) {
    int i = blockIdx.x * blockDim.x + threadIdx.x;
    if (i < n) p[i] = 0u;   // zero barrier counters + h0
}

__global__ __launch_bounds__(NTHREADS)
void esn_persistent_kernel(const float* __restrict__ x,
                           const float* __restrict__ wih,
                           const float* __restrict__ whh,
                           const float* __restrict__ bih,
                           const float* __restrict__ bhh,
                           float* __restrict__ out,
                           unsigned char* __restrict__ ws)
{
    extern __shared__ char smem[];
    __bf16* whh_l  = (__bf16*)(smem + WHH_OFF);   // [HS][HID] row-major
    __bf16* wih_l  = (__bf16*)(smem + WIH_OFF);   // [HS][IN_D]
    __bf16* h_l    = (__bf16*)(smem + HL_OFF);    // [BATCH][HID] staged h_{t-1}
    float*  bias_l = (float*) (smem + BIAS_OFF);  // fused bias slice

    unsigned* ctrs = (unsigned*)ws;
    __bf16*   hA   = (__bf16*)(ws + HA_OFF);
    __bf16*   hB   = (__bf16*)(ws + HB_OFF);

    const int tid   = threadIdx.x;
    const int wg    = blockIdx.x;
    const int jbase = wg * HS;

    // ---- one-time preload: weight slices f32 -> bf16 into LDS ----------
    for (int i = tid; i < HS * HID; i += NTHREADS) {
        int r = i >> 10, c = i & (HID - 1);
        whh_l[i] = (__bf16)whh[(size_t)(jbase + r) * HID + c];
    }
    for (int i = tid; i < HS * IN_D; i += NTHREADS) {
        int r = i >> 7, c = i & (IN_D - 1);
        wih_l[i] = (__bf16)wih[(jbase + r) * IN_D + c];
    }
    if (tid < HS) bias_l[tid] = bih[jbase + tid] + bhh[jbase + tid];
    __syncthreads();

    // ---- wave / lane geometry (wave32) ---------------------------------
    const int wave = tid >> 5;
    const int lane = tid & 31;
    const int l16  = lane & 15;
    const int lh   = lane >> 4;         // half-wave select
    const int b0   = (wave >> 2) * 16;  // tile row (batch)
    const int j0   = (wave & 3)  * 16;  // tile col (hidden, local)
    const int jg   = jbase + j0 + l16;  // global output column for this lane
    const int brow = b0 + l16;          // A-matrix row for this lane

    for (int t = 0; t < T_STEPS; ++t) {
        const __bf16* __restrict__ hcur = (t & 1) ? hB : hA;
        __bf16* __restrict__       hnxt = (t & 1) ? hA : hB;

        // ---- stage h_{t-1} (64x1024 bf16 = 128KB) into LDS via the
        //      CDNA5 async memory->LDS path (no VGPR round-trip,
        //      tracked by ASYNCcnt).  GV mode: per-lane 64-bit global
        //      address + per-lane 32-bit LDS byte address.
        for (int i = tid * 8; i < BATCH * HID; i += NTHREADS * 8) {
            const __bf16* src = hcur + i;
            unsigned lds_addr = (unsigned)(size_t)(h_l + i); // LDS aperture: low 32 bits
            asm volatile("global_load_async_to_lds_b128 %0, %1, off"
                         :
                         : "v"(lds_addr), "v"(src)
                         : "memory");
        }
        asm volatile("s_wait_asynccnt 0x0" ::: "memory");
        __syncthreads();

        v8f acc = {0.f, 0.f, 0.f, 0.f, 0.f, 0.f, 0.f, 0.f};

        // ---- recurrent GEMM: h_{t-1} @ Whh^T  (K = 1024) ---------------
        const __bf16* harow = h_l + brow * HID;
        const __bf16* wrow  = whh_l + (j0 + l16) * HID;
        #pragma unroll 4
        for (int kb = 0; kb < HID; kb += 32) {
            v8bf a0 = *(const v8bf*)(harow + kb + 16 * lh);
            v8bf a1 = *(const v8bf*)(harow + kb + 16 * lh + 8);
            v16bf a = __builtin_shufflevector(a0, a1,
                        0,1,2,3,4,5,6,7,8,9,10,11,12,13,14,15);
            v16bf b = *(const v16bf*)(wrow + kb + 16 * lh);
            acc = __builtin_amdgcn_wmma_f32_16x16x32_bf16(
                      false, a, false, b, (short)0, acc, false, false);
        }

        // ---- input GEMM: x_t @ Wih^T  (K = 128, f32 -> bf16 on the fly)
        const float* xrow = x + ((size_t)t * BATCH + brow) * IN_D;
        const __bf16* wirow = wih_l + (j0 + l16) * IN_D;
        #pragma unroll
        for (int kb = 0; kb < IN_D; kb += 32) {
            v4f f0 = *(const v4f*)(xrow + kb + 16 * lh);
            v4f f1 = *(const v4f*)(xrow + kb + 16 * lh + 4);
            v4f f2 = *(const v4f*)(xrow + kb + 16 * lh + 8);
            v4f f3 = *(const v4f*)(xrow + kb + 16 * lh + 12);
            v16bf a;
            #pragma unroll
            for (int e = 0; e < 4; ++e) {
                a[e]      = (__bf16)f0[e];
                a[e + 4]  = (__bf16)f1[e];
                a[e + 8]  = (__bf16)f2[e];
                a[e + 12] = (__bf16)f3[e];
            }
            v16bf b = *(const v16bf*)(wirow + kb + 16 * lh);
            acc = __builtin_amdgcn_wmma_f32_16x16x32_bf16(
                      false, a, false, b, (short)0, acc, false, false);
        }

        // prefetch next step's x rows into cache (global_prefetch_b8)
        if (t + 1 < T_STEPS)
            __builtin_prefetch(x + ((size_t)(t + 1) * BATCH + brow) * IN_D, 0, 1);

        // ---- bias + tanh + publish h_t ---------------------------------
        float bb = bias_l[j0 + l16];
        #pragma unroll
        for (int v = 0; v < 8; ++v) {
            int m = b0 + v + 8 * lh;                 // C/D layout: M = v + 8*lh
            float hv = tanhf(acc[v] + bb);
            hnxt[(size_t)m * HID + jg] = (__bf16)hv;
            if (t == T_STEPS - 1) out[(size_t)m * HID + jg] = hv;
        }

        // ---- device-wide barrier (per-step counter, no reset races) ----
        __threadfence();
        __syncthreads();
        if (tid == 0) {
            __hip_atomic_fetch_add(&ctrs[t], 1u, __ATOMIC_RELEASE,
                                   __HIP_MEMORY_SCOPE_AGENT);
            while (__hip_atomic_load(&ctrs[t], __ATOMIC_ACQUIRE,
                                     __HIP_MEMORY_SCOPE_AGENT) < NWG) {
                __builtin_amdgcn_s_sleep(1);
            }
        }
        __syncthreads();
    }
}

extern "C" void kernel_launch(void* const* d_in, const int* in_sizes, int n_in,
                              void* d_out, int out_size, void* d_ws, size_t ws_size,
                              hipStream_t stream) {
    const float* x   = (const float*)d_in[0];
    const float* wih = (const float*)d_in[1];
    const float* whh = (const float*)d_in[2];
    const float* bih = (const float*)d_in[3];
    const float* bhh = (const float*)d_in[4];
    float* out = (float*)d_out;
    unsigned char* ws = (unsigned char*)d_ws;

    // allow > 64KB dynamic LDS (WGP has 320KB)
    (void)hipFuncSetAttribute((const void*)esn_persistent_kernel,
                              hipFuncAttributeMaxDynamicSharedMemorySize,
                              LDS_BYTES);

    // zero barrier counters + h0 every call (deterministic under replay)
    int zero_words = (CTR_BYTES + BATCH * HID * 2) / 4;
    esn_init_kernel<<<(zero_words + 255) / 256, 256, 0, stream>>>(
        (unsigned*)ws, zero_words);

    esn_persistent_kernel<<<NWG, NTHREADS, LDS_BYTES, stream>>>(
        x, wih, whh, bih, bhh, out, ws);
}
